// SelfAttention_37117107372124
// MI455X (gfx1250) — matrix operand order repaired and checked by reading, not compiled, not run on hardware
//
#include <hip/hip_runtime.h>
#include <hip/hip_bf16.h>

typedef __bf16 bf16;
typedef __attribute__((ext_vector_type(4)))  __bf16 v4bf;
typedef __attribute__((ext_vector_type(8)))  __bf16 v8bf;
typedef __attribute__((ext_vector_type(16))) __bf16 v16bf;
typedef __attribute__((ext_vector_type(8)))  float  v8f;
typedef __attribute__((ext_vector_type(4)))  int    v4i;

// Pointee types for the async-to-LDS builtin (param1: AS1, param2: AS3).
typedef __attribute__((address_space(1))) v4i global_v4i;
typedef __attribute__((address_space(3))) v4i shared_v4i;

#define DIMX 2048
#define NH   16
#define HD   128
#define SLEN 2048
#define BATCH 2

#define HAS_ASYNC_LDS __has_builtin(__builtin_amdgcn_global_load_async_to_lds_b128)

static __device__ __forceinline__ v8f v8f_zero() {
    v8f z;
#pragma unroll
    for (int i = 0; i < 8; i++) z[i] = 0.0f;
    return z;
}

// Build a v16bf fragment from two 16-byte chunks (2x ds/global b128 loads).
static __device__ __forceinline__ v16bf frag16(const bf16* p0, const bf16* p1) {
    v8bf a = *(const v8bf*)p0;
    v8bf b = *(const v8bf*)p1;
    return __builtin_shufflevector(a, b, 0, 1, 2, 3, 4, 5, 6, 7,
                                   8, 9, 10, 11, 12, 13, 14, 15);
}

static __device__ __forceinline__ v8f wmma_bf16(v16bf a, v16bf b, v8f c) {
    return __builtin_amdgcn_wmma_f32_16x16x32_bf16(false, a, false, b,
                                                   (short)0, c, false, false);
}

// Async global->LDS b128 copy (CDNA5 ASYNCcnt path), with sync fallback.
static __device__ __forceinline__ void copy_b128_to_lds(const bf16* g, bf16* l) {
#if HAS_ASYNC_LDS
    __builtin_amdgcn_global_load_async_to_lds_b128(
        (global_v4i*)(v4i*)g, (shared_v4i*)(v4i*)l, 0, 0);
#else
    *(v8bf*)l = *(const v8bf*)g;
#endif
}

#if HAS_ASYNC_LDS
#if __has_builtin(__builtin_amdgcn_s_wait_asynccnt)
#define WAIT_ASYNC(n) __builtin_amdgcn_s_wait_asynccnt(n)
#else
#define WAIT_ASYNC(n) asm volatile("s_wait_asynccnt %0" ::"i"(n) : "memory")
#endif
#else
#define WAIT_ASYNC(n)
#endif

// ---------------------------------------------------------------------------
// Kernel 1: qkv = x @ W_qkv + b_qkv  (f32 in; bf16 out)
//   q,k stored [B,H,S,hd]; v stored TRANSPOSED [B,H,hd,S] so the attention
//   kernel can stage V^T in LDS with pure b128 copies (no element scatter).
// grid (48, 32), 256 threads. Block tile 128x128, wave tile 32x64, KT=32.
// ---------------------------------------------------------------------------
__global__ __launch_bounds__(256) void qkv_gemm_kernel(
    const float* __restrict__ X, const float* __restrict__ Wq,
    const float* __restrict__ bq, bf16* __restrict__ qo,
    bf16* __restrict__ ko, bf16* __restrict__ vo) {
    const int NALL = 3 * DIMX;  // 6144
    __shared__ bf16 Alds[128 * 40];  // [m][k], pad to 40
    __shared__ bf16 Blds[128 * 40];  // [n][k] (transposed), pad to 40

    const int tid = threadIdx.x;
    const int lane = tid & 31, wid = tid >> 5;
    const int wm = wid >> 1, wn = wid & 1;
    const int lh = lane & 15, hf = lane >> 4;
    const int mblk = blockIdx.y * 128, nblk = blockIdx.x * 128;

    v8f acc[2][4];
#pragma unroll
    for (int a = 0; a < 2; a++)
#pragma unroll
        for (int b = 0; b < 4; b++) acc[a][b] = v8f_zero();

    for (int kb = 0; kb < DIMX; kb += 32) {
        // A tile: 128x32 f32 -> bf16, row-major k-contiguous
#pragma unroll
        for (int i = 0; i < 4; i++) {
            int slot = tid + i * 256;
            int row = slot >> 3, c4 = slot & 7;
            const float4 f =
                *(const float4*)(X + (size_t)(mblk + row) * DIMX + kb + c4 * 4);
            v4bf w;
            w[0] = (bf16)f.x; w[1] = (bf16)f.y; w[2] = (bf16)f.z; w[3] = (bf16)f.w;
            *(v4bf*)&Alds[row * 40 + c4 * 4] = w;
        }
        // B tile: 32x128 f32 -> bf16, transposed into [n][k]
#pragma unroll
        for (int i = 0; i < 4; i++) {
            int slot = tid + i * 256;
            int row = slot >> 5, c4 = slot & 31;
            const float4 f = *(const float4*)(Wq + (size_t)(kb + row) * NALL +
                                              nblk + c4 * 4);
            Blds[(c4 * 4 + 0) * 40 + row] = (bf16)f.x;
            Blds[(c4 * 4 + 1) * 40 + row] = (bf16)f.y;
            Blds[(c4 * 4 + 2) * 40 + row] = (bf16)f.z;
            Blds[(c4 * 4 + 3) * 40 + row] = (bf16)f.w;
        }
        __syncthreads();

        v16bf af[2], bfr[4];
        const int abase = hf * 8;   // A: K halves at base, base+16
        const int bbase = hf * 8;   // B: K halves at base, base+16
#pragma unroll
        for (int mf = 0; mf < 2; mf++) {
            const bf16* p = &Alds[(wm * 32 + mf * 16 + lh) * 40 + abase];
            af[mf] = frag16(p, p + 16);
        }
#pragma unroll
        for (int nf = 0; nf < 4; nf++) {
            const bf16* p = &Blds[(wn * 64 + nf * 16 + lh) * 40 + bbase];
            bfr[nf] = frag16(p, p + 16);
        }
#pragma unroll
        for (int mf = 0; mf < 2; mf++)
#pragma unroll
            for (int nf = 0; nf < 4; nf++)
                acc[mf][nf] = wmma_bf16(af[mf], bfr[nf], acc[mf][nf]);
        __syncthreads();
    }

    // t is uniform per block (block spans 128 cols inside one 2048-col third)
    const int t = nblk >> 11;
    if (t < 2) {
        bf16* dst = (t == 0) ? qo : ko;  // [B,H,S,hd]
#pragma unroll
        for (int mf = 0; mf < 2; mf++)
#pragma unroll
            for (int nf = 0; nf < 4; nf++)
#pragma unroll
                for (int r = 0; r < 8; r++) {
                    int m = mblk + wm * 32 + mf * 16 + r + hf * 8;
                    int n = nblk + wn * 64 + nf * 16 + lh;
                    float val = acc[mf][nf][r] + bq[n];
                    int rem = n & 2047;
                    int h = rem >> 7, d = rem & 127;
                    int b = m >> 11, s = m & 2047;
                    dst[((size_t)((b * NH + h) * SLEN + s)) * HD + d] = (bf16)val;
                }
    } else {
        // v transposed: one packed b128 store per fragment (8 consecutive s)
#pragma unroll
        for (int mf = 0; mf < 2; mf++)
#pragma unroll
            for (int nf = 0; nf < 4; nf++) {
                int n = nblk + wn * 64 + nf * 16 + lh;
                int rem = n & 2047;
                int h = rem >> 7, d = rem & 127;
                int s0 = mblk + wm * 32 + mf * 16 + hf * 8;
                int b = s0 >> 11, s = s0 & 2047;
                v8bf pk;
#pragma unroll
                for (int r = 0; r < 8; r++)
                    pk[r] = (bf16)(acc[mf][nf][r] + bq[n]);
                *(v8bf*)&vo[((size_t)((b * NH + h) * HD + d)) * SLEN + s] = pk;
            }
    }
}

// ---------------------------------------------------------------------------
// Kernel 2: RoPE on q,k in-place ([B*H, S, 128] bf16). grid (512, 32), 256 thr.
// ---------------------------------------------------------------------------
__global__ __launch_bounds__(256) void rope_kernel(bf16* __restrict__ q,
                                                   bf16* __restrict__ k) {
    int bh = blockIdx.y;
    int s = blockIdx.x * 4 + (threadIdx.x >> 6);
    int j = threadIdx.x & 63;
    size_t base = ((size_t)bh * SLEN + s) * HD;
    // inv_freq[j] = 10000^{-j/64}
    float freq = __expf(-(float)j * (9.210340371976184f / 64.0f));
    float ang = (float)s * freq;
    float sn, cs;
    sincosf(ang, &sn, &cs);
    float q1 = (float)q[base + j], q2 = (float)q[base + j + 64];
    q[base + j]      = (bf16)(q1 * cs - q2 * sn);
    q[base + j + 64] = (bf16)(q2 * cs + q1 * sn);
    float k1 = (float)k[base + j], k2 = (float)k[base + j + 64];
    k[base + j]      = (bf16)(k1 * cs - k2 * sn);
    k[base + j + 64] = (bf16)(k2 * cs + k1 * sn);
}

// ---------------------------------------------------------------------------
// Kernel 3: flash attention per (b,h). grid (16, 32), 256 threads (8 waves).
// Double-buffered K/V^T LDS staging via async global->LDS (ASYNCcnt) so the
// next 64-key block streams from HBM while the current one runs 32 WMMAs/wave.
// v input is [B,H,hd,S] (pre-transposed). Output [B,S,D] bf16.
// ---------------------------------------------------------------------------
__global__ __launch_bounds__(256) void flash_kernel(
    const bf16* __restrict__ q, const bf16* __restrict__ k,
    const bf16* __restrict__ vt, bf16* __restrict__ o) {
    __shared__ bf16 Klds[2][64 * 136];    // [kpos][d], pad 136
    __shared__ bf16 Vlds[2][128 * 72];    // [d][kpos], pad 72
    __shared__ bf16 Plds[8 * 16 * 72];    // per-wave 16x64, pad 72

    const int tid = threadIdx.x;
    const int lane = tid & 31, wid = tid >> 5;
    const int lh = lane & 15, hf = lane >> 4;
    const int bh = blockIdx.y;
    const int q0 = blockIdx.x * 128 + wid * 16;
    const size_t bh_off = (size_t)bh * SLEN * HD;
    const float scale = 0.08838834764831845f;  // 1/sqrt(128)

    // Issue one 64-key block of K and V^T into LDS buffer `buf` (8 b128/thread)
    auto load_kv = [&](int buf, int kb) {
#pragma unroll
        for (int i = 0; i < 4; i++) {
            int chunk = tid + i * 256;
            int krow = chunk >> 4, kc = chunk & 15;  // K: 64 rows x 16 chunks
            copy_b128_to_lds(k + bh_off + (size_t)(kb + krow) * HD + kc * 8,
                             &Klds[buf][krow * 136 + kc * 8]);
            int vrow = chunk >> 3, vc = chunk & 7;   // V^T: 128 rows x 8 chunks
            copy_b128_to_lds(vt + bh_off + (size_t)vrow * SLEN + kb + vc * 8,
                             &Vlds[buf][vrow * 72 + vc * 8]);
        }
    };

    // Preload Q fragments (16 rows x 128) into registers
    v16bf aq[4];
    {
        const bf16* qp = q + bh_off + (size_t)(q0 + lh) * HD + hf * 8;
#pragma unroll
        for (int kf = 0; kf < 4; kf++)
            aq[kf] = frag16(qp + kf * 32, qp + kf * 32 + 16);
    }

    v8f oacc[8];
#pragma unroll
    for (int i = 0; i < 8; i++) oacc[i] = v8f_zero();
    float mrun[8], lrun[8];
#pragma unroll
    for (int r = 0; r < 8; r++) { mrun[r] = -INFINITY; lrun[r] = 0.0f; }

    bf16* pbuf = &Plds[wid * 16 * 72];
    const int NIT = SLEN / 64;  // 32

    load_kv(0, 0);
#pragma unroll 2
    for (int it = 0; it < NIT; ++it) {
        const int kb = it * 64;
        const int buf = it & 1;
        if (it > 0) __syncthreads();  // buffer we overwrite next is free
        if (it + 1 < NIT) {
            load_kv(buf ^ 1, kb + 64);
            WAIT_ASYNC(8);  // current block's 8 transfers done; next in flight
        } else {
            WAIT_ASYNC(0);
        }
        __syncthreads();

        // S = Q @ K^T (16 x 64)
        v8f sfr[4];
#pragma unroll
        for (int nf = 0; nf < 4; nf++) {
            v8f sa = v8f_zero();
#pragma unroll
            for (int kf = 0; kf < 4; kf++) {
                const bf16* p =
                    &Klds[buf][(nf * 16 + lh) * 136 + kf * 32 + hf * 8];
                sa = wmma_bf16(aq[kf], frag16(p, p + 16), sa);
            }
            sfr[nf] = sa;
        }
#pragma unroll
        for (int nf = 0; nf < 4; nf++)
#pragma unroll
            for (int r = 0; r < 8; r++) sfr[nf][r] *= scale;

        // Online softmax (row = r + 8*hf; cols live across 16 lanes)
        float mnew[8];
#pragma unroll
        for (int r = 0; r < 8; r++) {
            float mx = mrun[r];
#pragma unroll
            for (int nf = 0; nf < 4; nf++) mx = fmaxf(mx, sfr[nf][r]);
            mx = fmaxf(mx, __shfl_xor(mx, 1, 32));
            mx = fmaxf(mx, __shfl_xor(mx, 2, 32));
            mx = fmaxf(mx, __shfl_xor(mx, 4, 32));
            mx = fmaxf(mx, __shfl_xor(mx, 8, 32));
            mnew[r] = mx;
        }
        float psum[8];
#pragma unroll
        for (int r = 0; r < 8; r++) {
            float alpha = __expf(mrun[r] - mnew[r]);
            mrun[r] = mnew[r];
            lrun[r] *= alpha;
#pragma unroll
            for (int onf = 0; onf < 8; onf++) oacc[onf][r] *= alpha;
            psum[r] = 0.0f;
        }
#pragma unroll
        for (int nf = 0; nf < 4; nf++)
#pragma unroll
            for (int r = 0; r < 8; r++) {
                float p = __expf(sfr[nf][r] - mrun[r]);
                psum[r] += p;
                pbuf[(r + hf * 8) * 72 + nf * 16 + lh] = (bf16)p;
            }
#pragma unroll
        for (int r = 0; r < 8; r++) {
            float ps = psum[r];
            ps += __shfl_xor(ps, 1, 32);
            ps += __shfl_xor(ps, 2, 32);
            ps += __shfl_xor(ps, 4, 32);
            ps += __shfl_xor(ps, 8, 32);
            lrun[r] += ps;
        }

        // O += P @ V  (LDS ops in-order within a wave; pbuf is wave-private)
#pragma unroll
        for (int kf = 0; kf < 2; kf++) {
            const bf16* pp = &pbuf[lh * 72 + kf * 32 + hf * 8];
            v16bf ap = frag16(pp, pp + 16);
#pragma unroll
            for (int onf = 0; onf < 8; onf++) {
                const bf16* vp =
                    &Vlds[buf][(onf * 16 + lh) * 72 + kf * 32 + hf * 8];
                oacc[onf] = wmma_bf16(ap, frag16(vp, vp + 16), oacc[onf]);
            }
        }
    }

    // Epilogue: normalize by row sums, write [B,S,D] bf16
    const int b = bh >> 4, h = bh & 15;
#pragma unroll
    for (int r = 0; r < 8; r++) {
        float inv = 1.0f / lrun[r];
        int srow = q0 + r + hf * 8;
        size_t rowbase = ((size_t)(b * SLEN + srow)) * DIMX + h * HD;
#pragma unroll
        for (int onf = 0; onf < 8; onf++)
            o[rowbase + onf * 16 + lh] = (bf16)(oacc[onf][r] * inv);
    }
}

// ---------------------------------------------------------------------------
// Kernel 4: out = attn @ W_proj + b_proj  (bf16 A, f32 W->bf16, f32 out)
// grid (16, 32), 256 threads.
// ---------------------------------------------------------------------------
__global__ __launch_bounds__(256) void proj_gemm_kernel(
    const bf16* __restrict__ A, const float* __restrict__ W,
    const float* __restrict__ bias, float* __restrict__ out) {
    __shared__ bf16 Alds[128 * 40];
    __shared__ bf16 Blds[128 * 40];

    const int tid = threadIdx.x;
    const int lane = tid & 31, wid = tid >> 5;
    const int wm = wid >> 1, wn = wid & 1;
    const int lh = lane & 15, hf = lane >> 4;
    const int mblk = blockIdx.y * 128, nblk = blockIdx.x * 128;

    v8f acc[2][4];
#pragma unroll
    for (int a = 0; a < 2; a++)
#pragma unroll
        for (int b = 0; b < 4; b++) acc[a][b] = v8f_zero();

    for (int kb = 0; kb < DIMX; kb += 32) {
        // A tile: already bf16, 128x32 = 512 chunks of 8
#pragma unroll
        for (int i = 0; i < 2; i++) {
            int slot = tid + i * 256;
            int row = slot >> 2, c = slot & 3;
            v8bf a8 = *(const v8bf*)(A + (size_t)(mblk + row) * DIMX + kb + c * 8);
            *(v8bf*)&Alds[row * 40 + c * 8] = a8;
        }
        // B tile: 32x128 f32 -> bf16 transposed [n][k]
#pragma unroll
        for (int i = 0; i < 4; i++) {
            int slot = tid + i * 256;
            int row = slot >> 5, c4 = slot & 31;
            const float4 f = *(const float4*)(W + (size_t)(kb + row) * DIMX +
                                              nblk + c4 * 4);
            Blds[(c4 * 4 + 0) * 40 + row] = (bf16)f.x;
            Blds[(c4 * 4 + 1) * 40 + row] = (bf16)f.y;
            Blds[(c4 * 4 + 2) * 40 + row] = (bf16)f.z;
            Blds[(c4 * 4 + 3) * 40 + row] = (bf16)f.w;
        }
        __syncthreads();

        v16bf af[2], bfr[4];
        const int abase = hf * 8;
        const int bbase = hf * 8;
#pragma unroll
        for (int mf = 0; mf < 2; mf++) {
            const bf16* p = &Alds[(wm * 32 + mf * 16 + lh) * 40 + abase];
            af[mf] = frag16(p, p + 16);
        }
#pragma unroll
        for (int nf = 0; nf < 4; nf++) {
            const bf16* p = &Blds[(wn * 64 + nf * 16 + lh) * 40 + bbase];
            bfr[nf] = frag16(p, p + 16);
        }
#pragma unroll
        for (int mf = 0; mf < 2; mf++)
#pragma unroll
            for (int nf = 0; nf < 4; nf++)
                acc[mf][nf] = wmma_bf16(af[mf], bfr[nf], acc[mf][nf]);
        __syncthreads();
    }

#pragma unroll
    for (int mf = 0; mf < 2; mf++)
#pragma unroll
        for (int nf = 0; nf < 4; nf++)
#pragma unroll
            for (int r = 0; r < 8; r++) {
                int m = mblk + wm * 32 + mf * 16 + r + hf * 8;
                int n = nblk + wn * 64 + nf * 16 + lh;
                out[(size_t)m * DIMX + n] = acc[mf][nf][r] + bias[n];
            }
}

// ---------------------------------------------------------------------------
extern "C" void kernel_launch(void* const* d_in, const int* in_sizes, int n_in,
                              void* d_out, int out_size, void* d_ws,
                              size_t ws_size, hipStream_t stream) {
    (void)in_sizes; (void)n_in; (void)out_size; (void)ws_size;
    const float* x  = (const float*)d_in[0];
    const float* Wq = (const float*)d_in[1];
    const float* bq = (const float*)d_in[2];
    const float* Wp = (const float*)d_in[3];
    const float* bp = (const float*)d_in[4];
    float* out = (float*)d_out;

    const size_t TEN = (size_t)BATCH * NH * SLEN * HD;  // 8388608 elems
    bf16* q   = (bf16*)d_ws;           // [B,H,S,hd]
    bf16* k   = q + TEN;               // [B,H,S,hd]
    bf16* vt  = k + TEN;               // [B,H,hd,S] (transposed)
    bf16* att = vt + TEN;              // [B,S,D]; total 64 MB workspace

    qkv_gemm_kernel<<<dim3(48, 32), 256, 0, stream>>>(x, Wq, bq, q, k, vt);
    rope_kernel<<<dim3(512, 32), 256, 0, stream>>>(q, k);
    flash_kernel<<<dim3(16, 32), 256, 0, stream>>>(q, k, vt, att);
    proj_gemm_kernel<<<dim3(16, 32), 256, 0, stream>>>(att, Wp, bp, out);
}
